// Conceptual_Similarity_76819785056898
// MI455X (gfx1250) — matrix operand order repaired and checked by reading, not compiled, not run on hardware
//
#include <hip/hip_runtime.h>
#include <hip/hip_bf16.h>
#include <cstdint>

// ---------------- types ----------------
typedef __bf16 bf16;
typedef bf16  v16bf __attribute__((ext_vector_type(16)));
typedef bf16  v8bf  __attribute__((ext_vector_type(8)));
typedef float v8f   __attribute__((ext_vector_type(8)));

#define BB 64
#define SS 512
#define HB 768
#define HL 512
#define G4 2048   // 4*HL
#define TT 511
#define HC 512
#define CONCAT_DIM 2816  // HB + 4*HL

// ---------------- async copy to LDS (CDNA5) ----------------
#if defined(__HIP_DEVICE_COMPILE__) && __has_builtin(__builtin_amdgcn_global_load_async_to_lds_b128)
#define ASYNC_CP 1
typedef __attribute__((__vector_size__(4 * sizeof(int)))) int v4i_t;
#else
#define ASYNC_CP 0
#endif

__device__ __forceinline__ v16bf concat8(v8bf lo, v8bf hi) {
  return __builtin_shufflevector(lo, hi, 0,1,2,3,4,5,6,7,8,9,10,11,12,13,14,15);
}
__device__ __forceinline__ float sigmoidf_(float x) { return 1.f / (1.f + expf(-x)); }

// ============================================================
// 1) per-batch x/y + masked mean pools
// ============================================================
__global__ void prep_kernel(const float* __restrict__ bert,
                            const int* __restrict__ attn,
                            const int* __restrict__ seg,
                            float* __restrict__ s1_pool, float* __restrict__ s2_pool,
                            int* __restrict__ xv, int* __restrict__ yv)
{
  __shared__ int smin[256];
  __shared__ int ssum[256];
  __shared__ int sx, sy;
  const int b = blockIdx.x, tid = threadIdx.x;
  int mn = 1 << 30, sm = 0;
  for (int s = tid; s < SS; s += 256) {
    int a = attn[b * SS + s];
    sm += a;
    if (seg[b * SS + s] * a != 0 && s < mn) mn = s;
  }
  smin[tid] = mn; ssum[tid] = sm;
  __syncthreads();
  for (int off = 128; off > 0; off >>= 1) {
    if (tid < off) {
      smin[tid] = min(smin[tid], smin[tid + off]);
      ssum[tid] += ssum[tid + off];
    }
    __syncthreads();
  }
  if (tid == 0) {
    int first = (smin[0] >= (1 << 30)) ? 0 : smin[0];
    sx = first - 1;          // x = argmax(seg*mask) - 1
    sy = ssum[0] - 1;        // y = sum(mask) - 1
    xv[b] = sx; yv[b] = sy;
  }
  __syncthreads();
  const int x = sx, y = sy;
  const float c1 = fmaxf((float)(x - 1), 1.f);
  const float c2 = fmaxf((float)(y - x - 1), 1.f);
  for (int h = tid; h < HB; h += 256) {
    float a1 = 0.f, a2 = 0.f;
    for (int s = 0; s < SS; ++s) {
      float v = bert[((size_t)b * SS + s) * HB + h];
      if (s >= 1 && s < x) a1 += v;
      if (s > x  && s < y) a2 += v;
    }
    s1_pool[b * HB + h] = a1 / c1;
    s2_pool[b * HB + h] = a2 / c2;
  }
}

// ============================================================
// 2) build s1_seq / s2_seq in bf16, row = t*64+b
// ============================================================
__global__ void build_seq_kernel(const float* __restrict__ bert,
                                 const float* __restrict__ s1_pool,
                                 const float* __restrict__ s2_pool,
                                 const int* __restrict__ xv,
                                 bf16* __restrict__ s1q, bf16* __restrict__ s2q)
{
  const int blk = blockIdx.x;          // t*64 + b
  const int b = blk & 63, t = blk >> 6;
  const int xx = xv[b];
  int i2 = xx + 1 + t;
  i2 = i2 < 0 ? 0 : (i2 > SS - 1 ? SS - 1 : i2);
  for (int h = threadIdx.x; h < HB; h += 256) {
    float v1 = bert[((size_t)b * SS + (1 + t)) * HB + h] * s2_pool[b * HB + h];
    float v2 = bert[((size_t)b * SS + i2) * HB + h] * s1_pool[b * HB + h];
    s1q[(size_t)blk * HB + h] = (bf16)v1;
    s2q[(size_t)blk * HB + h] = (bf16)v2;
  }
}

// ============================================================
// 3) f32 -> bf16 convert
// ============================================================
__global__ void f32_to_bf16_kernel(const float* __restrict__ src, bf16* __restrict__ dst, int n)
{
  int i = blockIdx.x * 256 + threadIdx.x;
  if (i < n) dst[i] = (bf16)src[i];
}

// ============================================================
// 4) bf16 WMMA GEMM: C[M,N] = A[M,K] * B[N,K]^T  (+bias,relu f32 out)
//    block = 256 threads (8 waves), C-tile 64x128, A staged in LDS
// ============================================================
template <bool RELU_F32_OUT>
__global__ void wmma_gemm_bf16_kernel(const bf16* __restrict__ A,
                                      const bf16* __restrict__ Bm,
                                      void* __restrict__ Cout,
                                      const float* __restrict__ bias,
                                      int M, int N, int K)
{
  __shared__ bf16 As[64 * 32];
  const int tid  = threadIdx.x;
  const int m0   = blockIdx.y * 64;
  const int n0   = blockIdx.x * 128;
  const int wave = tid >> 5, lane = tid & 31;
  const int wy = wave & 3, wx = wave >> 2;
  const int lm = lane & 15, hiHalf = lane >> 4;

  v8f acc[4] = {};

  const int row = tid >> 2, ch = tid & 3;   // 64 rows x 4 chunks of 16B
  const int Ksteps = K >> 5;
  for (int kk = 0; kk < Ksteps; ++kk) {
    const bf16* src = A + (size_t)(m0 + row) * K + kk * 32 + ch * 8;
    bf16* dst = &As[row * 32 + ch * 8];
#if ASYNC_CP
    __builtin_amdgcn_global_load_async_to_lds_b128(
        (v4i_t*)src, (v4i_t*)dst, 0, 0);
    asm volatile("s_wait_asynccnt 0" ::: "memory");
#else
    *(v8bf*)dst = *(const v8bf*)src;
#endif
    __syncthreads();

    const bf16* ap = &As[(wy * 16 + lm) * 32 + (hiHalf << 4)];
    v16bf a = concat8(*(const v8bf*)ap, *(const v8bf*)(ap + 8));
    const int kb = kk * 32 + (hiHalf << 4);
#pragma unroll
    for (int nt = 0; nt < 4; ++nt) {
      const bf16* bp = Bm + (size_t)(n0 + wx * 64 + nt * 16 + lm) * K + kb;
      v16bf bfr = *(const v16bf*)bp;
      acc[nt] = __builtin_amdgcn_wmma_f32_16x16x32_bf16(
          false, a, false, bfr, (short)0, acc[nt], false, false);
    }
    __syncthreads();
  }

  const int mb = hiHalf << 3;
#pragma unroll
  for (int nt = 0; nt < 4; ++nt) {
    const int n = n0 + wx * 64 + nt * 16 + lm;
#pragma unroll
    for (int r = 0; r < 8; ++r) {
      const int m = m0 + wy * 16 + mb + r;
      float v = acc[nt][r];
      if (RELU_F32_OUT) {
        v = fmaxf(v + bias[n], 0.f);
        ((float*)Cout)[(size_t)m * N + n] = v;
      } else {
        ((bf16*)Cout)[(size_t)m * N + n] = (bf16)v;
      }
    }
  }
}

// ============================================================
// 5) masked LSTM scan, one workgroup (32 waves) per (seq,dir)
//    h (bf16, double-buffered) + c (f32) live in LDS (256 KB)
//    per-step gates = h @ Whh^T via WMMA, + xp(t) + biases
// ============================================================
__global__ void __launch_bounds__(1024)
lstm_scan_kernel(const bf16* __restrict__ xp,    // [T*64, 2048]
                 const bf16* __restrict__ Whh,   // [2048, 512]
                 const float* __restrict__ b_ih, const float* __restrict__ b_hh,
                 const float* __restrict__ h0, const float* __restrict__ c0, // [2,64,512]
                 const int* __restrict__ xv, const int* __restrict__ yv,
                 float* __restrict__ h_out,      // [64,512]
                 int seq_id, int dir)
{
  extern __shared__ char smem_raw[];
  bf16*  hS = (bf16*)smem_raw;                              // [2][64*512]
  float* cS = (float*)(smem_raw + 2 * BB * HL * sizeof(bf16)); // [64*512]
  const int tid = threadIdx.x;

  for (int i = tid; i < BB * HL; i += 1024) {
    hS[i] = (bf16)h0[dir * BB * HL + i];
    hS[BB * HL + i] = (bf16)0.f;
    cS[i] = c0[dir * BB * HL + i];
  }
  __syncthreads();

  const int wave = tid >> 5, lane = tid & 31;
  const int mt = wave >> 3, ug = wave & 7;
  const int mrow0 = mt * 16;
  const int lm = lane & 15, hiHalf = lane >> 4;
  int cur = 0;

  for (int step = 0; step < TT; ++step) {
    const int t = dir ? (TT - 1 - step) : step;
    const bf16* hR = hS + cur * (BB * HL);
    bf16* hW = hS + (cur ^ 1) * (BB * HL);

    for (int nt = 0; nt < 4; ++nt) {
      const int u0 = ug * 64 + nt * 16;
      v8f acc[4] = {};
      for (int kk = 0; kk < HL / 32; ++kk) {
        const int k0 = kk * 32 + (hiHalf << 4);
        const bf16* hp = hR + (mrow0 + lm) * HL + k0;
        v16bf a = concat8(*(const v8bf*)hp, *(const v8bf*)(hp + 8));
        const int kb = kk * 32 + (hiHalf << 4);
#pragma unroll
        for (int g = 0; g < 4; ++g) {
          const bf16* bp = Whh + (size_t)(g * HL + u0 + lm) * HL + kb;
          v16bf bfr = *(const v16bf*)bp;
          acc[g] = __builtin_amdgcn_wmma_f32_16x16x32_bf16(
              false, a, false, bfr, (short)0, acc[g], false, false);
        }
      }
      const int u  = u0 + lm;
      const int mb = hiHalf << 3;
      const float bi = b_ih[u]            + b_hh[u];
      const float bf = b_ih[HL + u]       + b_hh[HL + u];
      const float bg = b_ih[2 * HL + u]   + b_hh[2 * HL + u];
      const float bo = b_ih[3 * HL + u]   + b_hh[3 * HL + u];
#pragma unroll
      for (int r = 0; r < 8; ++r) {
        const int b = mrow0 + mb + r;
        const size_t xrow = ((size_t)t * BB + b) * G4;
        float gi = acc[0][r] + (float)xp[xrow + u]            + bi;
        float gf = acc[1][r] + (float)xp[xrow + HL + u]       + bf;
        float gg = acc[2][r] + (float)xp[xrow + 2 * HL + u]   + bg;
        float go = acc[3][r] + (float)xp[xrow + 3 * HL + u]   + bo;
        const int idx = b * HL + u;
        float cprev = cS[idx];
        float hprev = (float)hR[idx];
        float cn = sigmoidf_(gf) * cprev + sigmoidf_(gi) * tanhf(gg);
        float hn = sigmoidf_(go) * tanhf(cn);
        bool valid = (seq_id == 0) ? (t < xv[b] - 1) : (t < yv[b] - xv[b] - 1);
        cS[idx] = valid ? cn : cprev;
        hW[idx] = valid ? (bf16)hn : (bf16)hprev;
      }
    }
    __syncthreads();
    cur ^= 1;
  }

  for (int i = tid; i < BB * HL; i += 1024)
    h_out[i] = (float)hS[cur * (BB * HL) + i];
}

// ============================================================
// 6) concat [cls, s1_hf, s1_hb, s2_hf, s2_hb] -> bf16
// ============================================================
__global__ void concat_kernel(const float* __restrict__ bert,
                              const float* __restrict__ h_final, // [4][64][512]
                              bf16* __restrict__ cq)              // [64][2816]
{
  const int b = blockIdx.x;
  for (int j = threadIdx.x; j < CONCAT_DIM; j += 256) {
    float v;
    if (j < HB) {
      v = bert[(size_t)b * SS * HB + j];     // token 0
    } else {
      int k = j - HB;
      int c = k >> 9, u = k & 511;
      v = h_final[((size_t)c * BB + b) * HL + u];
    }
    cq[(size_t)b * CONCAT_DIM + j] = (bf16)v;
  }
}

// ============================================================
// 7) out = sigmoid(hid @ W2^T + b2)
// ============================================================
__global__ void out_kernel(const float* __restrict__ hid,
                           const float* __restrict__ W2,
                           const float* __restrict__ b2,
                           float* __restrict__ out)
{
  __shared__ float red[256];
  const int b = blockIdx.x, tid = threadIdx.x;
  float s = 0.f;
  for (int u = tid; u < HC; u += 256) s += hid[b * HC + u] * W2[u];
  red[tid] = s;
  __syncthreads();
  for (int off = 128; off > 0; off >>= 1) {
    if (tid < off) red[tid] += red[tid + off];
    __syncthreads();
  }
  if (tid == 0) out[b] = 1.f / (1.f + expf(-(red[0] + b2[0])));
}

// ============================================================
// launch
// ============================================================
extern "C" void kernel_launch(void* const* d_in, const int* in_sizes, int n_in,
                              void* d_out, int out_size, void* d_ws, size_t ws_size,
                              hipStream_t stream)
{
  const float* bert   = (const float*)d_in[0];
  const int*   attn   = (const int*)d_in[1];
  const int*   seg    = (const int*)d_in[2];
  const float* h0     = (const float*)d_in[3];
  const float* c0     = (const float*)d_in[4];
  const float* W_ih_f = (const float*)d_in[5];
  const float* W_hh_f = (const float*)d_in[6];
  const float* b_ih_f = (const float*)d_in[7];
  const float* b_hh_f = (const float*)d_in[8];
  const float* W_ih_b = (const float*)d_in[9];
  const float* W_hh_b = (const float*)d_in[10];
  const float* b_ih_b = (const float*)d_in[11];
  const float* b_hh_b = (const float*)d_in[12];
  const float* W1     = (const float*)d_in[13];
  const float* b1     = (const float*)d_in[14];
  const float* W2     = (const float*)d_in[15];
  const float* b2     = (const float*)d_in[16];
  float* out = (float*)d_out;

  char* ws = (char*)d_ws;
  size_t off = 0;
  auto alloc = [&](size_t bytes) -> char* {
    char* p = ws + off;
    off += (bytes + 255) & ~(size_t)255;
    return p;
  };

  bf16* s1q    = (bf16*)alloc((size_t)TT * BB * HB * 2);
  bf16* s2q    = (bf16*)alloc((size_t)TT * BB * HB * 2);
  bf16* wihf_q = (bf16*)alloc((size_t)G4 * HB * 2);
  bf16* wihb_q = (bf16*)alloc((size_t)G4 * HB * 2);
  bf16* whhf_q = (bf16*)alloc((size_t)G4 * HL * 2);
  bf16* whhb_q = (bf16*)alloc((size_t)G4 * HL * 2);
  bf16* w1q    = (bf16*)alloc((size_t)HC * CONCAT_DIM * 2);
  bf16* xp_q   = (bf16*)alloc((size_t)TT * BB * G4 * 2);     // reused per combo
  float* s1_pool = (float*)alloc((size_t)BB * HB * 4);
  float* s2_pool = (float*)alloc((size_t)BB * HB * 4);
  int*  xvv    = (int*)alloc(BB * 4);
  int*  yvv    = (int*)alloc(BB * 4);
  float* h_fin = (float*)alloc((size_t)4 * BB * HL * 4);
  bf16* cq     = (bf16*)alloc((size_t)BB * CONCAT_DIM * 2);
  float* hid   = (float*)alloc((size_t)BB * HC * 4);

  prep_kernel<<<BB, 256, 0, stream>>>(bert, attn, seg, s1_pool, s2_pool, xvv, yvv);
  build_seq_kernel<<<TT * BB, 256, 0, stream>>>(bert, s1_pool, s2_pool, xvv, s1q, s2q);

  auto cvt = [&](const float* s, bf16* d, int n) {
    f32_to_bf16_kernel<<<(n + 255) / 256, 256, 0, stream>>>(s, d, n);
  };
  cvt(W_ih_f, wihf_q, G4 * HB);
  cvt(W_ih_b, wihb_q, G4 * HB);
  cvt(W_hh_f, whhf_q, G4 * HL);
  cvt(W_hh_b, whhb_q, G4 * HL);
  cvt(W1, w1q, HC * CONCAT_DIM);

  const size_t scan_lds = 2 * (size_t)BB * HL * sizeof(bf16) + (size_t)BB * HL * sizeof(float);
  for (int combo = 0; combo < 4; ++combo) {
    const int seq_id = combo >> 1, dir = combo & 1;
    const bf16* Aseq = seq_id ? s2q : s1q;
    const bf16* Wih  = dir ? wihb_q : wihf_q;
    const bf16* Whh  = dir ? whhb_q : whhf_q;
    const float* bih = dir ? b_ih_b : b_ih_f;
    const float* bhh = dir ? b_hh_b : b_hh_f;

    // xp = seq @ Wih^T  : M = 511*64 = 32704, K = 768, N = 2048
    wmma_gemm_bf16_kernel<false><<<dim3(G4 / 128, (TT * BB) / 64), 256, 0, stream>>>(
        Aseq, Wih, (void*)xp_q, nullptr, TT * BB, G4, HB);

    lstm_scan_kernel<<<1, 1024, scan_lds, stream>>>(
        xp_q, Whh, bih, bhh, h0, c0, xvv, yvv,
        h_fin + (size_t)combo * BB * HL, seq_id, dir);
  }

  concat_kernel<<<BB, 256, 0, stream>>>(bert, h_fin, cq);

  // hid = relu(concat @ W1^T + b1) : M=64, K=2816, N=512
  wmma_gemm_bf16_kernel<true><<<dim3(HC / 128, BB / 64), 256, 0, stream>>>(
      cq, w1q, (void*)hid, b1, BB, HC, CONCAT_DIM);

  out_kernel<<<BB, 256, 0, stream>>>(hid, W2, b2, out);
}
